// RoutingCapsule_69217692942526
// MI455X (gfx1250) — matrix operand order repaired and checked by reading, not compiled, not run on hardware
//
#include <hip/hip_runtime.h>

typedef __attribute__((ext_vector_type(16))) _Float16 v16h;
typedef __attribute__((ext_vector_type(8)))  float    v8f;
typedef __attribute__((ext_vector_type(4)))  float    f32x4;

#define BSZ  128
#define PDIM 2048   // CL*H*W = 8*16*16
#define NPK  32     // primary capsule length (GEMM K)
#define KK   160    // NC*CLO = 10*16 (GEMM N)
#define NC   10
#define CLO  16

// ---------------- squash(tensor) along NP, cast to f16 ----------------
__global__ void rc_squash_x(const float* __restrict__ t, _Float16* __restrict__ xh) {
    int idx = blockIdx.x * blockDim.x + threadIdx.x;   // (b*P + p)
    if (idx >= BSZ * PDIM) return;
    const f32x4* src = reinterpret_cast<const f32x4*>(t + (size_t)idx * NPK);
    float vals[NPK];
    float ss = 0.f;
    for (int c = 0; c < 8; ++c) {
        f32x4 v = src[c];
        for (int r = 0; r < 4; ++r) { float f = v[r]; vals[c*4+r] = f; ss += f*f; }
    }
    float sc = sqrtf(ss) / (1.0f + ss);                // (ss/(1+ss))/sqrt(ss)
    union { f32x4 f4; _Float16 h[8]; } o;
    f32x4* dst = reinterpret_cast<f32x4*>(xh + (size_t)idx * NPK);
    for (int c = 0; c < 4; ++c) {
        for (int r = 0; r < 8; ++r) o.h[r] = (_Float16)(vals[c*8+r] * sc);
        dst[c] = o.f4;
    }
}

// ---------------- weight (P,32,160) f32 -> wT[p][n][k] f16 ----------------
__global__ void rc_castw(const float* __restrict__ w, _Float16* __restrict__ wT) {
    int t = blockIdx.x * blockDim.x + threadIdx.x;
    if (t >= PDIM * KK) return;
    int n = t % KK, p = t / KK;
    union { f32x4 f4[4]; _Float16 h[NPK]; } o;
    for (int k = 0; k < NPK; ++k)                       // coalesced across n
        o.h[k] = (_Float16)w[((size_t)p * NPK + k) * KK + n];
    f32x4* dst = reinterpret_cast<f32x4*>(wT + ((size_t)p * KK + n) * NPK);
    for (int c = 0; c < 4; ++c) dst[c] = o.f4[c];
}

// ---------------- generic zero ----------------
__global__ void rc_zero(float* p, int n) {
    for (int i = blockIdx.x * blockDim.x + threadIdx.x; i < n; i += gridDim.x * blockDim.x)
        p[i] = 0.f;
}

// ---------------- u[b][p][kk] = x[b][p][:] @ W[p][:][kk]  via WMMA ----------------
// A = W^T tile (M=kk 16, K=32), B = x tile (K=32, N=batch 16), D f32 -> f16
__global__ __launch_bounds__(320) void rc_gemm_u(const _Float16* __restrict__ xh,
                                                 const _Float16* __restrict__ wT,
                                                 _Float16* __restrict__ u) {
    __shared__ __align__(16) _Float16 lx[BSZ * NPK];    // x[:, p, :]  (8 KB)
    const int p   = blockIdx.x;
    const int tid = threadIdx.x;
    // stage x slice: 128 rows * 64B = 512 x 16B chunks
    for (int idx = tid; idx < 512; idx += 320) {
        int row = idx >> 2, c4 = idx & 3;
        reinterpret_cast<f32x4*>(lx + row * NPK)[c4] =
            reinterpret_cast<const f32x4*>(xh + ((size_t)row * PDIM + p) * NPK)[c4];
    }
    __syncthreads();

    const int wv   = tid >> 5;          // wave -> kk tile (0..9)
    const int lane = tid & 31;
    const int lo   = lane & 15;
    const bool hi  = lane >= 16;
    const int kk0  = wv * 16;

    // A (16x32 f16): lane<16 rows M=lane hold K{0..7,16..23}; lane>=16 hold K{8..15,24..31}
    union { v16h v; f32x4 f4[2]; } A;
    const _Float16* abase = wT + ((size_t)p * KK + kk0 + lo) * NPK;
    const int a0 = hi ? 8 : 0;
    A.f4[0] = *reinterpret_cast<const f32x4*>(abase + a0);
    A.f4[1] = *reinterpret_cast<const f32x4*>(abase + a0 + 16);

    const int b0h = hi ? 8 : 0;         // B: lanes 0-15 K{0..7,16..23}, lanes 16-31 K{8..15,24..31}
    for (int bt = 0; bt < 8; ++bt) {
        union { v16h v; f32x4 f4[2]; } Bm;
        const _Float16* bbase = lx + (bt * 16 + lo) * NPK + b0h;
        Bm.f4[0] = *reinterpret_cast<const f32x4*>(bbase);
        Bm.f4[1] = *reinterpret_cast<const f32x4*>(bbase + 16);

        v8f c = {0.f,0.f,0.f,0.f,0.f,0.f,0.f,0.f};
        c = __builtin_amdgcn_wmma_f32_16x16x32_f16(false, A.v, false, Bm.v,
                                                   (short)0, c, false, false);
        // D: N=batch in lane, M=kk = kk0 + (hi?8:0) + r  -> contiguous per-lane store
        union { f32x4 f4; _Float16 h[8]; } D;
        for (int r = 0; r < 8; ++r) D.h[r] = (_Float16)c[r];
        const int b   = bt * 16 + lo;
        const int kks = kk0 + (hi ? 8 : 0);
        *reinterpret_cast<f32x4*>(u + ((size_t)b * PDIM + p) * KK + kks) = D.f4;
    }
}

// ---------------- one routing pass: (bias += u.v) -> softmax -> s += c*u ----------------
__global__ __launch_bounds__(320) void rc_route(const _Float16* __restrict__ u,
                                                float* __restrict__ bias,
                                                const float* __restrict__ vprev,
                                                float* __restrict__ s,
                                                int addBias) {
    __shared__ float c_lds[256 * NC];
    __shared__ float v_lds[KK];
    const int b   = blockIdx.x;
    const int p0  = blockIdx.y * 256;
    const int tid = threadIdx.x;

    if (addBias && tid < KK) v_lds[tid] = vprev[b * KK + tid];
    __syncthreads();

    if (tid < 256) {
        const int p = p0 + tid;
        float* bptr = bias + ((size_t)b * PDIM + p) * NC;
        float bl[NC];
        for (int n = 0; n < NC; ++n) bl[n] = bptr[n];
        const _Float16* up = u + ((size_t)b * PDIM + p) * KK;
        if (addBias) {
            for (int n = 0; n < NC; ++n) {
                float d = 0.f;
                for (int k = 0; k < CLO; ++k) d += (float)up[n*CLO + k] * v_lds[n*CLO + k];
                bl[n] += d;
            }
            for (int n = 0; n < NC; ++n) bptr[n] = bl[n];
        }
        float m = bl[0];
        for (int n = 1; n < NC; ++n) m = fmaxf(m, bl[n]);
        float e[NC], sum = 0.f;
        for (int n = 0; n < NC; ++n) { e[n] = __expf(bl[n] - m); sum += e[n]; }
        float inv = 1.f / sum;
        for (int n = 0; n < NC; ++n) c_lds[tid * NC + n] = e[n] * inv;
    }
    __syncthreads();

    const int n    = tid >> 5;          // wave -> output capsule
    const int lane = tid & 31;
    const int k    = lane & 15;
    const int h    = lane >> 4;
    float acc = 0.f;
    for (int jj = h; jj < 256; jj += 2) {
        const int p = p0 + jj;
        acc += c_lds[jj * NC + n] * (float)u[((size_t)b * PDIM + p) * KK + n*CLO + k];
    }
    acc += __shfl_xor(acc, 16, 32);
    if (h == 0) atomicAdd(&s[(b * NC + n) * CLO + k], acc);
}

// ---------------- v = squash(s) along CLO; write v_ws and d_out ----------------
__global__ void rc_squash_v(const float* __restrict__ s, float* __restrict__ vws,
                            float* __restrict__ out) {
    int t = blockIdx.x * blockDim.x + threadIdx.x;     // (b*NC + n)
    if (t >= BSZ * NC) return;
    const float* sp = s + t * CLO;
    float vals[CLO], ss = 0.f;
    for (int k = 0; k < CLO; ++k) { vals[k] = sp[k]; ss += vals[k]*vals[k]; }
    float sc = sqrtf(ss) / (1.f + ss);
    for (int k = 0; k < CLO; ++k) {
        float r = vals[k] * sc;
        vws[t * CLO + k] = r;
        out[t * CLO + k] = r;
    }
}

extern "C" void kernel_launch(void* const* d_in, const int* in_sizes, int n_in,
                              void* d_out, int out_size, void* d_ws, size_t ws_size,
                              hipStream_t stream) {
    const float* tensor = (const float*)d_in[0];
    const float* weight = (const float*)d_in[1];
    float* out = (float*)d_out;
    char* ws = (char*)d_ws;

    _Float16* xh   = (_Float16*)(ws);                   // 128*2048*32*2   = 16,777,216
    _Float16* wT   = (_Float16*)(ws + 16777216);        // 2048*160*32*2   = 20,971,520
    _Float16* u    = (_Float16*)(ws + 37748736);        // 128*2048*160*2  = 83,886,080
    float*    bias = (float*)(ws + 121634816);          // 128*2048*10*4   = 10,485,760
    float*    s    = (float*)(ws + 132120576);          // 128*10*16*4     = 81,920
    float*    vws  = (float*)(ws + 132202496);          // 81,920

    rc_squash_x<<<(BSZ*PDIM + 255) / 256, 256, 0, stream>>>(tensor, xh);
    rc_castw<<<(PDIM*KK + 255) / 256, 256, 0, stream>>>(weight, wT);
    rc_zero<<<1024, 256, 0, stream>>>(bias, BSZ * PDIM * NC);
    rc_gemm_u<<<PDIM, 320, 0, stream>>>(xh, wT, u);

    for (int i = 0; i < 3; ++i) {
        rc_zero<<<80, 256, 0, stream>>>(s, BSZ * NC * CLO);
        dim3 g(BSZ, PDIM / 256);
        rc_route<<<g, 320, 0, stream>>>(u, bias, vws, s, i > 0 ? 1 : 0);
        rc_squash_v<<<(BSZ*NC + 127) / 128, 128, 0, stream>>>(s, vws, out);
    }
}
